// S4Model_21990232555643
// MI455X (gfx1250) — matrix operand order repaired and checked by reading, not compiled, not run on hardware
//
#include <hip/hip_runtime.h>
#include <hip/hip_bf16.h>
#include <math.h>

// ---------------- problem constants ----------------
#define B_    16
#define L_    8192
#define DIN_  128
#define H_    256
#define NHALF_ 32
#define HO_   512          // 2H
#define NC_   256          // chunks per sequence (L / 32)
// ---------------------------------------------------

typedef __bf16 bf16_t;
typedef __attribute__((ext_vector_type(16))) __bf16 v16bf;
typedef __attribute__((ext_vector_type(8)))  float  v8f;

union Frag16 { uint4 q[2]; v16bf v; };

// A-matrix K index for lane l, element i (contiguous k order used for A and B alike in this file)
__device__ __forceinline__ int ka_idx(int lane, int i) {
  return ((lane >> 4) * 16) + i;
}

// =============== weight pre-pack into WMMA B-fragment order ===============
// packed index p = ((kc*NT + nt)*32 + lane)*16 + i
// lane: N = lane&15 ; K = (lane>>4)*16 + i  (contiguous k order, same as the A side)
__global__ __launch_bounds__(256) void k_pack(const float* __restrict__ W,
                                              bf16_t* __restrict__ pk,
                                              int K, int Ncols) {
  int t = blockIdx.x * blockDim.x + threadIdx.x;
  if (t >= K * Ncols) return;
  int i    = t & 15;
  int lane = (t >> 4) & 31;
  int rest = t >> 9;                 // kc*NT + nt
  int NT   = Ncols >> 4;
  int nt   = rest % NT;
  int kc   = rest / NT;
  int krow = kc * 32 + ((lane >> 4) * 16) + i;
  int n    = nt * 16 + (lane & 15);
  pk[t] = (bf16_t)W[krow * Ncols + n];
}

// =============== S4D discrete coefficients ===============
// cf layout (floats): [0]=w_re [8192]=w_im [16384]=cc_re [24576]=cc_im
//                     [32768]=w32_re [40960]=w32_im
__global__ __launch_bounds__(256) void k_coef(const float* __restrict__ log_dt,
                                              const float* __restrict__ log_A_real,
                                              const float* __restrict__ A_imag,
                                              const float* __restrict__ C_re,
                                              const float* __restrict__ C_im,
                                              float* __restrict__ cf) {
  int t = blockIdx.x * blockDim.x + threadIdx.x;
  if (t >= H_ * NHALF_) return;
  int h = t >> 5;
  float dt  = expf(log_dt[h]);
  float ar  = -expf(log_A_real[t]);
  float ai  = A_imag[t];
  float dre = ar * dt, dim = ai * dt;
  float em  = expf(dre);
  float w_re = em * cosf(dim), w_im = em * sinf(dim);
  float n_re = w_re - 1.f, n_im = w_im;
  float den  = ar * ar + ai * ai;
  float q_re = (n_re * ar + n_im * ai) / den;
  float q_im = (n_im * ar - n_re * ai) / den;
  float cr = C_re[t], ci = C_im[t];
  cf[t]         = w_re;
  cf[8192 + t]  = w_im;
  cf[16384 + t] = cr * q_re - ci * q_im;
  cf[24576 + t] = cr * q_im + ci * q_re;
}

// power table: wpow[(h*32+n)*33 + d] = w^d (re,im interleaved); also w^32 -> cf
__global__ __launch_bounds__(256) void k_wpow(float* __restrict__ cf,
                                              float* __restrict__ wpow) {
  int t = blockIdx.x * blockDim.x + threadIdx.x;
  if (t >= H_ * NHALF_) return;
  float wr = cf[t], wi = cf[8192 + t];
  float pr = 1.f, pi = 0.f;
  for (int d = 0; d <= 32; ++d) {
    wpow[(t * 33 + d) * 2]     = pr;
    wpow[(t * 33 + d) * 2 + 1] = pi;
    if (d == 32) { cf[32768 + t] = pr; cf[40960 + t] = pi; }
    float nr = pr * wr - pi * wi;
    float ni = pr * wi + pi * wr;
    pr = nr; pi = ni;
  }
}

// k[h][d] = 2 * sum_n Re(cc_n * w_n^d), d = 0..32
__global__ __launch_bounds__(256) void k_kvec(const float* __restrict__ cf,
                                              const float* __restrict__ wpow,
                                              float* __restrict__ kvec) {
  int t = blockIdx.x * blockDim.x + threadIdx.x;
  if (t >= H_ * 33) return;
  int h = t / 33, d = t % 33;
  float s = 0.f;
  for (int n = 0; n < NHALF_; ++n) {
    int ci = h * 32 + n;
    int wi = ((ci) * 33 + d) * 2;
    s += cf[16384 + ci] * wpow[wi] - cf[24576 + ci] * wpow[wi + 1];
  }
  kvec[t] = 2.f * s;
}

// T (32x32 Toeplitz, A-frag order): pkT[((h*2+mt)*32+lane)*16+i]
__global__ __launch_bounds__(256) void k_packT(const float* __restrict__ kvec,
                                               bf16_t* __restrict__ pkT) {
  int t = blockIdx.x * blockDim.x + threadIdx.x;
  if (t >= H_ * 2 * 32 * 16) return;
  int i = t & 15, lane = (t >> 4) & 31, mt = (t >> 9) & 1, h = t >> 10;
  int j  = mt * 16 + (lane & 15);
  int kk = ka_idx(lane, i);
  pkT[t] = (j >= kk) ? (bf16_t)kvec[h * 33 + (j - kk)] : (bf16_t)0.f;
}

// V (64x32, rows 0..31 = Re(w^{31-i}) per state, 32..63 = Im): pkV[((h*4+mt)*32+lane)*16+i]
__global__ __launch_bounds__(256) void k_packV(const float* __restrict__ wpow,
                                               bf16_t* __restrict__ pkV) {
  int t = blockIdx.x * blockDim.x + threadIdx.x;
  if (t >= H_ * 4 * 32 * 16) return;
  int i = t & 15, lane = (t >> 4) & 31, mt = (t >> 9) & 3, h = t >> 11;
  int comp = mt * 16 + (lane & 15);
  int n = comp & 31, im = comp >> 5;
  int kk = ka_idx(lane, i);
  pkV[t] = (bf16_t)wpow[((h * 32 + n) * 33 + (31 - kk)) * 2 + im];
}

// M2 (32x64): y_carry[j] = sum_n 2Re(cc w^{j+1}) s_re - 2Im(cc w^{j+1}) s_im
// pkM2[(((h*2+kc)*2+mt)*32+lane)*16+i]
__global__ __launch_bounds__(256) void k_packM2(const float* __restrict__ cf,
                                                const float* __restrict__ wpow,
                                                bf16_t* __restrict__ pkM2) {
  int t = blockIdx.x * blockDim.x + threadIdx.x;
  if (t >= H_ * 2 * 2 * 32 * 16) return;
  int i = t & 15, lane = (t >> 4) & 31, mt = (t >> 9) & 1, kc = (t >> 10) & 1, h = t >> 11;
  int j  = mt * 16 + (lane & 15);
  int kk = kc * 32 + ka_idx(lane, i);
  int n = kk & 31, im = kk >> 5;
  int ci = h * 32 + n;
  int wi = (ci * 33 + (j + 1)) * 2;
  float pr = wpow[wi], pi = wpow[wi + 1];
  float ccr = cf[16384 + ci], cci = cf[24576 + ci];
  float val = (im == 0) ? 2.f * (ccr * pr - cci * pi)
                        : -2.f * (ccr * pi + cci * pr);
  pkM2[t] = (bf16_t)val;
}

__global__ void k_pool_init(unsigned* __restrict__ pool) {
  int t = blockIdx.x * blockDim.x + threadIdx.x;
  if (t < B_ * H_) pool[t] = 0x007FFFFFu;   // monotone encoding of -inf
}

// =============== fc1 + ReLU + LayerNorm  (WMMA bf16) ===============
__global__ __launch_bounds__(256) void k_fc1_ln(const float* __restrict__ x,
                                                const float* __restrict__ b1,
                                                const float* __restrict__ ln_g,
                                                const float* __restrict__ ln_b,
                                                const bf16_t* __restrict__ pkW1,
                                                bf16_t* __restrict__ uT) {
  __shared__ bf16_t xs[16][136];
  __shared__ float  hs[16][264];
  __shared__ float  red[2][16][16];
  __shared__ float  mu_s[16], rs_s[16];

  const int tid = threadIdx.x, blk = blockIdx.x;
  const int b  = blk >> 9;
  const int t0 = (blk & 511) << 4;
  const long rowbase = (long)blk * 16;

  {
    int t = tid >> 4, seg = tid & 15;
    const float* src = x + (rowbase + t) * DIN_ + seg * 8;
    float4 f0 = ((const float4*)src)[0];
    float4 f1 = ((const float4*)src)[1];
    alignas(16) bf16_t tv[8];
    tv[0]=(bf16_t)f0.x; tv[1]=(bf16_t)f0.y; tv[2]=(bf16_t)f0.z; tv[3]=(bf16_t)f0.w;
    tv[4]=(bf16_t)f1.x; tv[5]=(bf16_t)f1.y; tv[6]=(bf16_t)f1.z; tv[7]=(bf16_t)f1.w;
    *(uint4*)&xs[t][seg * 8] = *(const uint4*)tv;
  }
  __syncthreads();

  const int lane = tid & 31, w = tid >> 5;
  const int trow = lane & 15;
  const int kb   = (lane >> 4) * 16;

  v8f acc[2] = {{}, {}};
  for (int kc = 0; kc < 4; ++kc) {
    Frag16 a;
    a.q[0] = *(const uint4*)&xs[trow][kc * 32 + kb];
    a.q[1] = *(const uint4*)&xs[trow][kc * 32 + kb + 8];
    #pragma unroll
    for (int j = 0; j < 2; ++j) {
      int nt = w * 2 + j;
      Frag16 bw;
      const uint4* bp = (const uint4*)&pkW1[(((kc * 16) + nt) * 32 + lane) * 16];
      bw.q[0] = bp[0]; bw.q[1] = bp[1];
      acc[j] = __builtin_amdgcn_wmma_f32_16x16x32_bf16(
          false, a.v, false, bw.v, (short)0, acc[j], false, false);
    }
  }

  #pragma unroll
  for (int j = 0; j < 2; ++j) {
    int col = (w * 2 + j) * 16 + (lane & 15);
    float bias = b1[col];
    #pragma unroll
    for (int r = 0; r < 8; ++r) {
      int m = r + ((lane >> 4) * 8);
      hs[m][col] = fmaxf(acc[j][r] + bias, 0.f);
    }
  }
  __syncthreads();

  {
    int r = tid >> 4, seg = tid & 15;
    float s = 0.f, s2 = 0.f;
    #pragma unroll
    for (int i = 0; i < 16; ++i) { float v = hs[r][seg * 16 + i]; s += v; s2 += v * v; }
    red[0][r][seg] = s; red[1][r][seg] = s2;
  }
  __syncthreads();
  if (tid < 16) {
    float s = 0.f, s2 = 0.f;
    #pragma unroll
    for (int i = 0; i < 16; ++i) { s += red[0][tid][i]; s2 += red[1][tid][i]; }
    float mu = s * (1.f / H_);
    float var = s2 * (1.f / H_) - mu * mu;
    mu_s[tid] = mu;
    rs_s[tid] = rsqrtf(var + 1e-5f);
  }
  __syncthreads();

  {
    int h = tid;
    float g = ln_g[h], bb = ln_b[h];
    alignas(16) bf16_t outv[16];
    #pragma unroll
    for (int t = 0; t < 16; ++t)
      outv[t] = (bf16_t)((hs[t][h] - mu_s[t]) * rs_s[t] * g + bb);
    bf16_t* dst = uT + ((long)b * H_ + h) * (long)L_ + t0;
    ((uint4*)dst)[0] = ((const uint4*)outv)[0];
    ((uint4*)dst)[1] = ((const uint4*)outv)[1];
  }
}

// =============== S4D stage 1: Z1 = T@U, VS = V@U  (WMMA) ===============
// one block per (b,h); 8 waves x 2 chunk-tiles cover N=256 chunks
__global__ __launch_bounds__(256) void k_stage1(const bf16_t* __restrict__ uT,
                                                const bf16_t* __restrict__ pkT,
                                                const bf16_t* __restrict__ pkV,
                                                bf16_t* __restrict__ z1,
                                                bf16_t* __restrict__ S) {
  __shared__ bf16_t ystage[8][1024];
  const int bh = blockIdx.x;
  const int h  = bh & 255;
  const int lane = threadIdx.x & 31, w = threadIdx.x >> 5;
  const bf16_t* ub = uT + (long)bh * L_;

  Frag16 aT[2], aV[4];
  #pragma unroll
  for (int mt = 0; mt < 2; ++mt) {
    const uint4* p = (const uint4*)&pkT[((h * 2 + mt) * 32 + lane) * 16];
    aT[mt].q[0] = p[0]; aT[mt].q[1] = p[1];
  }
  #pragma unroll
  for (int mt = 0; mt < 4; ++mt) {
    const uint4* p = (const uint4*)&pkV[((h * 4 + mt) * 32 + lane) * 16];
    aV[mt].q[0] = p[0]; aV[mt].q[1] = p[1];
  }

  #pragma unroll
  for (int jn = 0; jn < 2; ++jn) {
    int nt = w * 2 + jn;
    Frag16 bu;   // U B-frag: K = time-in-chunk, N = chunk
    const uint4* up = (const uint4*)&ub[(nt * 16 + (lane & 15)) * 32 + (lane >> 4) * 16];
    bu.q[0] = up[0]; bu.q[1] = up[1];

    v8f zt[2] = {{}, {}};
    v8f vt[4] = {{}, {}, {}, {}};
    #pragma unroll
    for (int mt = 0; mt < 2; ++mt)
      zt[mt] = __builtin_amdgcn_wmma_f32_16x16x32_bf16(
          false, aT[mt].v, false, bu.v, (short)0, zt[mt], false, false);
    #pragma unroll
    for (int mt = 0; mt < 4; ++mt)
      vt[mt] = __builtin_amdgcn_wmma_f32_16x16x32_bf16(
          false, aV[mt].v, false, bu.v, (short)0, vt[mt], false, false);

    // VS -> S (bf16, layout [bh][c][64], contiguous 16B per lane per mtile)
    long cbase = ((long)bh * NC_ + nt * 16 + (lane & 15)) * 64 + 8 * (lane >> 4);
    #pragma unroll
    for (int mt = 0; mt < 4; ++mt) {
      alignas(16) bf16_t sv[8];
      #pragma unroll
      for (int r = 0; r < 8; ++r) sv[r] = (bf16_t)vt[mt][r];
      *(uint4*)&S[cbase + mt * 16] = *(const uint4*)sv;
    }
    // Z1 -> wave-private LDS slice for re-coalescing
    #pragma unroll
    for (int mt = 0; mt < 2; ++mt)
      #pragma unroll
      for (int r = 0; r < 8; ++r)
        ystage[w][(jn * 16 + (lane & 15)) * 32 + mt * 16 + r + 8 * (lane >> 4)] =
            (bf16_t)zt[mt][r];
  }

  // coalesced copy LDS -> z1 (64B per lane)
  {
    alignas(16) bf16_t tv[32];
    #pragma unroll
    for (int q = 0; q < 4; ++q)
      ((uint4*)tv)[q] = *(const uint4*)&ystage[w][lane * 32 + q * 8];
    bf16_t* dst = z1 + (long)bh * L_ + w * 1024 + lane * 32;
    #pragma unroll
    for (int q = 0; q < 4; ++q) ((uint4*)dst)[q] = ((const uint4*)tv)[q];
  }
}

// =============== S4D stage 2: chunk-boundary state chain (cheap) ===============
// one wave per (b,h); lane = state n; rewrites S[c] with the carry-IN of chunk c
__global__ __launch_bounds__(256) void k_chain(const float* __restrict__ cf,
                                               bf16_t* __restrict__ S) {
  const int lane = threadIdx.x & 31;
  const int wid  = blockIdx.x * 8 + (threadIdx.x >> 5);   // bh
  const int h = wid & 255;
  const float w32r = cf[32768 + h * 32 + lane];
  const float w32i = cf[40960 + h * 32 + lane];
  bf16_t* Sp = S + (long)wid * NC_ * 64;
  float sr = 0.f, si = 0.f;
  for (int c = 0; c < NC_; ++c) {
    if (c < NC_ - 8) __builtin_prefetch(&Sp[(c + 8) * 64 + lane], 0, 0);
    float vr = (float)Sp[c * 64 + lane];
    float vi = (float)Sp[c * 64 + 32 + lane];
    Sp[c * 64 + lane]      = (bf16_t)sr;   // carry-in for chunk c
    Sp[c * 64 + 32 + lane] = (bf16_t)si;
    float nr = fmaf(w32r, sr, fmaf(-w32i, si, vr));
    float ni = fmaf(w32r, si, fmaf(w32i, sr, vi));
    sr = nr; si = ni;
  }
}

// =============== S4D stage 3: y = Z1 + M2@S_carry + Dp*u, GELU (WMMA) ===============
__global__ __launch_bounds__(256) void k_stage3(const bf16_t* __restrict__ uT,
                                                const bf16_t* __restrict__ S,
                                                const bf16_t* __restrict__ pkM2,
                                                const float* __restrict__ Dp,
                                                bf16_t* __restrict__ y) {
  __shared__ bf16_t ystage[8][1024];
  const int bh = blockIdx.x;
  const int h  = bh & 255;
  const int lane = threadIdx.x & 31, w = threadIdx.x >> 5;
  const float dp = Dp[h];

  Frag16 aM[2][2];   // [kc][mt]
  #pragma unroll
  for (int kc = 0; kc < 2; ++kc)
    #pragma unroll
    for (int mt = 0; mt < 2; ++mt) {
      const uint4* p = (const uint4*)&pkM2[(((h * 2 + kc) * 2 + mt) * 32 + lane) * 16];
      aM[kc][mt].q[0] = p[0]; aM[kc][mt].q[1] = p[1];
    }

  #pragma unroll
  for (int jn = 0; jn < 2; ++jn) {
    int nt = w * 2 + jn;
    v8f acc[2] = {{}, {}};
    #pragma unroll
    for (int kc = 0; kc < 2; ++kc) {
      Frag16 bs;   // S B-frag: K = state comp, N = chunk
      const uint4* sp = (const uint4*)
          &S[((long)bh * NC_ + nt * 16 + (lane & 15)) * 64 + kc * 32 + (lane >> 4) * 16];
      bs.q[0] = sp[0]; bs.q[1] = sp[1];
      #pragma unroll
      for (int mt = 0; mt < 2; ++mt)
        acc[mt] = __builtin_amdgcn_wmma_f32_16x16x32_bf16(
            false, aM[kc][mt].v, false, bs.v, (short)0, acc[mt], false, false);
    }
    #pragma unroll
    for (int mt = 0; mt < 2; ++mt)
      #pragma unroll
      for (int r = 0; r < 8; ++r)
        ystage[w][(jn * 16 + (lane & 15)) * 32 + mt * 16 + r + 8 * (lane >> 4)] =
            (bf16_t)acc[mt][r];
  }

  // final fuse: y = z1 + carry + dp*u, exact GELU (in-place, wave-private span)
  {
    long base = (long)bh * L_ + w * 1024 + lane * 32;
    alignas(16) bf16_t zu[32], uu[32], cc[32], outv[32];
    #pragma unroll
    for (int q = 0; q < 4; ++q) {
      ((uint4*)zu)[q] = ((const uint4*)(y  + base))[q];
      ((uint4*)uu)[q] = ((const uint4*)(uT + base))[q];
      ((uint4*)cc)[q] = *(const uint4*)&ystage[w][lane * 32 + q * 8];
    }
    #pragma unroll
    for (int k = 0; k < 32; ++k) {
      float yv = (float)zu[k] + (float)cc[k] + dp * (float)uu[k];
      float g  = 0.5f * yv * (1.f + erff(yv * 0.70710678118f));
      outv[k] = (bf16_t)g;
    }
    #pragma unroll
    for (int q = 0; q < 4; ++q) ((uint4*)(y + base))[q] = ((const uint4*)outv)[q];
  }
}

// =============== y @ Wo + GLU + max-pool over L  (WMMA bf16) ===============
__global__ __launch_bounds__(256) void k_glu_pool(const bf16_t* __restrict__ yT,
                                                  const bf16_t* __restrict__ pkWo,
                                                  const float* __restrict__ bo,
                                                  unsigned* __restrict__ pool) {
  __shared__ bf16_t As[16][264];

  const int tid = threadIdx.x, blk = blockIdx.x;
  const int b  = blk >> 9;
  const int t0 = (blk & 511) << 4;

  {
    int h = tid;
    alignas(16) bf16_t tv[16];
    const bf16_t* src = yT + ((long)b * H_ + h) * (long)L_ + t0;
    ((uint4*)tv)[0] = ((const uint4*)src)[0];
    ((uint4*)tv)[1] = ((const uint4*)src)[1];
    #pragma unroll
    for (int t = 0; t < 16; ++t) As[t][h] = tv[t];
  }
  __syncthreads();

  const int lane = tid & 31, w = tid >> 5;
  const int trow = lane & 15;
  const int kb   = (lane >> 4) * 16;
  const int nts[4] = {2 * w, 2 * w + 1, 2 * w + 16, 2 * w + 17};

  v8f acc[4] = {{}, {}, {}, {}};
  for (int kc = 0; kc < 8; ++kc) {
    Frag16 a;
    a.q[0] = *(const uint4*)&As[trow][kc * 32 + kb];
    a.q[1] = *(const uint4*)&As[trow][kc * 32 + kb + 8];
    #pragma unroll
    for (int j = 0; j < 4; ++j) {
      Frag16 bw;
      const uint4* bp = (const uint4*)&pkWo[((kc * 32 + nts[j]) * 32 + lane) * 16];
      bw.q[0] = bp[0]; bw.q[1] = bp[1];
      acc[j] = __builtin_amdgcn_wmma_f32_16x16x32_bf16(
          false, a.v, false, bw.v, (short)0, acc[j], false, false);
    }
  }

  #pragma unroll
  for (int j = 0; j < 2; ++j) {
    int colA = nts[j] * 16 + (lane & 15);
    int colB = colA + H_;
    float ba = bo[colA], bb = bo[colB];
    float m = -INFINITY;
    #pragma unroll
    for (int r = 0; r < 8; ++r) {
      float a  = acc[j][r] + ba;
      float gt = acc[j + 2][r] + bb;
      float gl = a / (1.f + expf(-gt));
      m = fmaxf(m, gl);
    }
    m = fmaxf(m, __shfl_xor(m, 16, 32));
    if (lane < 16) {
      unsigned bits = __float_as_uint(m);
      unsigned enc = (bits & 0x80000000u) ? ~bits : (bits | 0x80000000u);
      atomicMax(&pool[b * H_ + colA], enc);
    }
  }
}

// =============== head ===============
__global__ __launch_bounds__(256) void k_head(const unsigned* __restrict__ pool,
                                              const float* __restrict__ W2,
                                              const float* __restrict__ b2,
                                              const float* __restrict__ W3,
                                              const float* __restrict__ b3,
                                              float* __restrict__ out) {
  __shared__ float ps[B_][H_];
  __shared__ float fs[B_][512];
  __shared__ float lg[B_][10];
  const int tid = threadIdx.x;

  for (int idx = tid; idx < B_ * H_; idx += 256) {
    unsigned e = pool[idx];
    unsigned bits = (e & 0x80000000u) ? (e & 0x7FFFFFFFu) : ~e;
    ps[idx >> 8][idx & 255] = __uint_as_float(bits);
  }
  __syncthreads();

  for (int idx = tid; idx < B_ * 512; idx += 256) {
    int b = idx >> 9, j = idx & 511;
    float s = b2[j];
    for (int h = 0; h < H_; ++h) s = fmaf(ps[b][h], W2[h * 512 + j], s);
    fs[b][j] = (s > 0.f) ? s : 0.01f * s;
  }
  __syncthreads();

  if (tid < 160) {
    int b = tid / 10, c = tid % 10;
    float s = b3[c];
    for (int j = 0; j < 512; ++j) s = fmaf(fs[b][j], W3[j * 10 + c], s);
    lg[b][c] = s;
    out[b * 10 + c] = s;
  }
  __syncthreads();
  if (tid < 10) {
    int best = 0; float bv = lg[0][tid];
    for (int b = 1; b < B_; ++b)
      if (lg[b][tid] > bv) { bv = lg[b][tid]; best = b; }
    out[160 + tid] = (float)best;
  }
}

// =========================== host launcher ===========================
static inline size_t alignup(size_t x) { return (x + 255) & ~(size_t)255; }

extern "C" void kernel_launch(void* const* d_in, const int* in_sizes, int n_in,
                              void* d_out, int out_size, void* d_ws, size_t ws_size,
                              hipStream_t stream) {
  const float* x          = (const float*)d_in[0];
  const float* W1         = (const float*)d_in[1];
  const float* b1         = (const float*)d_in[2];
  const float* ln_g       = (const float*)d_in[3];
  const float* ln_b       = (const float*)d_in[4];
  const float* log_dt     = (const float*)d_in[5];
  const float* log_A_real = (const float*)d_in[6];
  const float* A_imag     = (const float*)d_in[7];
  const float* C_re       = (const float*)d_in[8];
  const float* C_im       = (const float*)d_in[9];
  const float* Dp         = (const float*)d_in[10];
  const float* Wo         = (const float*)d_in[11];
  const float* bo         = (const float*)d_in[12];
  const float* W2         = (const float*)d_in[13];
  const float* b2         = (const float*)d_in[14];
  const float* W3         = (const float*)d_in[15];
  const float* b3         = (const float*)d_in[16];
  float* out = (float*)d_out;

  char* ws = (char*)d_ws;
  size_t off = 0;
  bf16_t* uT   = (bf16_t*)(ws + off); off += alignup((size_t)B_*H_*L_*2);        // 64MB
  bf16_t* yT   = (bf16_t*)(ws + off); off += alignup((size_t)B_*H_*L_*2);        // 64MB
  bf16_t* S    = (bf16_t*)(ws + off); off += alignup((size_t)B_*H_*NC_*64*2);    // 128MB
  float*  cf   = (float*) (ws + off); off += alignup((size_t)6*8192*4);
  float*  wpow = (float*) (ws + off); off += alignup((size_t)8192*33*2*4);
  float*  kvec = (float*) (ws + off); off += alignup((size_t)H_*33*4);
  bf16_t* pkW1 = (bf16_t*)(ws + off); off += alignup((size_t)DIN_*H_*2);
  bf16_t* pkWo = (bf16_t*)(ws + off); off += alignup((size_t)H_*HO_*2);
  bf16_t* pkT  = (bf16_t*)(ws + off); off += alignup((size_t)H_*2*32*16*2);
  bf16_t* pkV  = (bf16_t*)(ws + off); off += alignup((size_t)H_*4*32*16*2);
  bf16_t* pkM2 = (bf16_t*)(ws + off); off += alignup((size_t)H_*2*2*32*16*2);
  unsigned* pool = (unsigned*)(ws + off); off += alignup((size_t)B_*H_*4);

  // ---- setup: weight packs, SSM matrices, pool init ----
  hipLaunchKernelGGL(k_pack,   dim3(128),  dim3(256), 0, stream, W1, pkW1, DIN_, H_);
  hipLaunchKernelGGL(k_pack,   dim3(512),  dim3(256), 0, stream, Wo, pkWo, H_, HO_);
  hipLaunchKernelGGL(k_coef,   dim3(32),   dim3(256), 0, stream,
                     log_dt, log_A_real, A_imag, C_re, C_im, cf);
  hipLaunchKernelGGL(k_wpow,   dim3(32),   dim3(256), 0, stream, cf, wpow);
  hipLaunchKernelGGL(k_kvec,   dim3(33),   dim3(256), 0, stream, cf, wpow, kvec);
  hipLaunchKernelGGL(k_packT,  dim3(1024), dim3(256), 0, stream, kvec, pkT);
  hipLaunchKernelGGL(k_packV,  dim3(2048), dim3(256), 0, stream, wpow, pkV);
  hipLaunchKernelGGL(k_packM2, dim3(2048), dim3(256), 0, stream, cf, wpow, pkM2);
  hipLaunchKernelGGL(k_pool_init, dim3(16), dim3(256), 0, stream, pool);

  // ---- main pipeline ----
  hipLaunchKernelGGL(k_fc1_ln,   dim3(8192), dim3(256), 0, stream,
                     x, b1, ln_g, ln_b, pkW1, uT);
  hipLaunchKernelGGL(k_stage1,   dim3(4096), dim3(256), 0, stream, uT, pkT, pkV, yT, S);
  hipLaunchKernelGGL(k_chain,    dim3(512),  dim3(256), 0, stream, cf, S);
  hipLaunchKernelGGL(k_stage3,   dim3(4096), dim3(256), 0, stream, uT, S, pkM2, Dp, yT);
  hipLaunchKernelGGL(k_glu_pool, dim3(8192), dim3(256), 0, stream, yT, pkWo, bo, pool);
  hipLaunchKernelGGL(k_head,     dim3(1),    dim3(256), 0, stream,
                     pool, W2, b2, W3, b3, out);
}
